// HyperCNN_67551245631519
// MI455X (gfx1250) — matrix operand order repaired and checked by reading, not compiled, not run on hardware
//
#include <hip/hip_runtime.h>

// ---------------- problem constants ----------------
#define B_   16
#define T_   8
#define K_   64
#define CO_  128
#define CI_  128
#define H_   64
#define W_   64
#define NTAP 25                 // 5x5 taps
#define NCOL 409600             // CO*CI*NTAP (hypernet GEMM N)
#define KVOL 3200               // CI*NTAP   (per-o weight volume)
#define NCHUNK 100              // NTAP * (CI/32) K-chunks in the conv

// ---------------- WMMA types ----------------
typedef __attribute__((ext_vector_type(16))) _Float16 v16h;
typedef __attribute__((ext_vector_type(8)))  float    v8f;

union Frag16 { v16h v; uint4 q[2]; };

// ---------------- f32 -> f16 pair pack ----------------
static __device__ inline unsigned pack_f16(float a, float b) {
    union { _Float16 h[2]; unsigned u; } p;
    p.h[0] = (_Float16)a; p.h[1] = (_Float16)b;
    return p.u;
}

// =====================================================================
// Kernel 1: coeff[b][k] = (1/T) * sum_t tf[b][t][k]          (tiny)
// =====================================================================
__global__ __launch_bounds__(256) void k_coeff(const float* __restrict__ tf,
                                               float* __restrict__ coeff) {
    const int i = blockIdx.x * 256 + threadIdx.x;     // 0..1023
    const int b = i >> 6, k = i & 63;
    float s = 0.f;
#pragma unroll
    for (int t = 0; t < T_; ++t) s += tf[(b * T_ + t) * K_ + k];
    coeff[i] = s * (1.0f / (float)T_);
}

// =====================================================================
// Kernel 2: lp16[b][n] = f16( sum_k coeff[b][k] * W[k][n] )
// n = o*3200 + c*25 + ij (natural W layout). Bandwidth-bound: one
// coalesced pass over the 105 MB weight bank.
// =====================================================================
__global__ __launch_bounds__(256) void k_lp_gemm(const float* __restrict__ wbank,
                                                 const float* __restrict__ coeff,
                                                 _Float16* __restrict__ lp16) {
    __shared__ float sc[B_ * K_];
    for (int i = threadIdx.x; i < B_ * K_; i += 256) sc[i] = coeff[i];
    __syncthreads();

    const int n = blockIdx.x * 256 + threadIdx.x;     // < 409600 (1600 blocks)
    float acc[B_] = {};
    for (int k = 0; k < K_; ++k) {
        const float wv = wbank[(size_t)k * NCOL + n];
#pragma unroll
        for (int b = 0; b < B_; ++b) acc[b] += sc[b * K_ + k] * wv;
    }
#pragma unroll
    for (int b = 0; b < B_; ++b)
        lp16[(size_t)b * NCOL + n] = (_Float16)acc[b];
}

// =====================================================================
// Kernel 3: permute lp16 [b][o][c][ij] -> lpp [b][ij][o][c] so the conv
// K-dim (channels at fixed tap) is contiguous. LDS-staged transpose.
// =====================================================================
__global__ __launch_bounds__(256) void k_permute(const unsigned short* __restrict__ lp16,
                                                 unsigned short* __restrict__ lpp) {
    __shared__ unsigned short s[KVOL];
    const int b = blockIdx.x >> 7;
    const int o = blockIdx.x & 127;
    const unsigned short* src = lp16 + ((size_t)b * CO_ + o) * KVOL;
    for (int i = threadIdx.x; i < KVOL; i += 256) s[i] = src[i];
    __syncthreads();

    const int c2 = (threadIdx.x & 63) * 2;            // even channel
#pragma unroll
    for (int pass = 0; pass < 7; ++pass) {
        const int ij = pass * 4 + (threadIdx.x >> 6);
        if (ij < 25) {
            const unsigned u = (unsigned)s[c2 * NTAP + ij] |
                               ((unsigned)s[(c2 + 1) * NTAP + ij] << 16);
            *(unsigned*)((char*)lpp +
                (((((size_t)(b * NTAP + ij)) * CO_ + o) * CI_ + c2) * 2)) = u;
        }
    }
}

// =====================================================================
// Kernel 4: implicit-GEMM conv, double-buffered LDS, async A staging.
// Block = (image b, output row h0): 128 o x 64 w, 8 waves (4o x 2p),
// wave = 32o x 32p = 4 fragments; 100 K-chunks of 32 channels.
//   stage(q+1) (async A -> LDS + B gather/convert) overlaps WMMA(q).
// =====================================================================
#define LDSP   80                         // tile row pitch (32 f16 + 16B pad)
#define ATILE  (128 * LDSP)               // 10240 B
#define BTILE  (64 * LDSP)                // 5120 B
#define BUFSZ  (ATILE + BTILE)            // 15360 B per buffer

__global__ __launch_bounds__(256) void k_conv_wmma(const float* __restrict__ x,
                                                   const unsigned short* __restrict__ lpp,
                                                   float* __restrict__ y) {
    __shared__ __align__(16) unsigned char smem[2 * BUFSZ];

    const int t  = threadIdx.x;
    const int b  = blockIdx.x >> 6;
    const int h0 = blockIdx.x & 63;

    const int wv     = t >> 5;
    const int lane   = t & 31;
    const int lrow   = lane & 15;
    const int lhi    = lane >> 4;
    const int o_wave = (wv & 3) * 32;
    const int p_wave = (wv >> 2) * 32;

    // staging thread mappings (chunk-invariant)
    const int ar = t >> 1, apart = t & 1;   // A: 2 threads per 64B row
    const int bw = t & 63, bcp = t >> 6;    // B: pixel w, channel-pair group

    v8f acc[2][2] = {};

    auto stage = [&](int q, int buf) {
        const int ij = q >> 2, c0 = (q & 3) * 32;
        const int di = ij / 5, dj = ij % 5;
        const int hh = h0 + di - 2;

        // ---- A tile: lpp[b][ij][o][c0..c0+32) -> LDS, async (2 x b128) ----
        {
            const size_t eoff =
                (size_t)((b * NTAP + ij) * CO_ + ar) * CI_ + c0 + apart * 16;
            const unsigned long long ga =
                (unsigned long long)(uintptr_t)(lpp + eoff);
            const unsigned la =
                (unsigned)(uintptr_t)(smem + buf * BUFSZ + ar * LDSP + apart * 32);
            asm volatile("global_load_async_to_lds_b128 %0, %1, off"
                         :: "v"(la), "v"(ga) : "memory");
            asm volatile("global_load_async_to_lds_b128 %0, %1, off"
                         :: "v"(la + 16u), "v"(ga + 16ull) : "memory");
            // warm L2/WGP$ one tap ahead for the async engine
            if (ij + 1 < NTAP)
                __builtin_prefetch((const void*)(lpp + eoff + (size_t)CO_ * CI_), 0, 1);
        }

        // ---- B tile: x[b][c][hh][w+dj-2] -> LDS [w][c] f16, halo/pad = 0 ----
        {
            const int  ww = bw + dj - 2;
            const bool ok = (ww >= 0) && (ww < W_) && (hh >= 0) && (hh < H_);
            unsigned char* bb = smem + buf * BUFSZ + ATILE + bw * LDSP;
#pragma unroll
            for (int pass = 0; pass < 4; ++pass) {
                const int ce = (bcp + pass * 4) * 2;    // even channel 0..30
                float f0 = 0.f, f1 = 0.f;
                if (ok) {
                    const size_t base =
                        ((size_t)(b * CI_ + c0 + ce) * H_ + hh) * W_ + ww;
                    f0 = x[base];
                    f1 = x[base + (size_t)H_ * W_];
                }
                *(unsigned*)(bb + ce * 2) = pack_f16(f0, f1);
            }
        }
    };

    stage(0, 0);

    for (int q = 0; q < NCHUNK; ++q) {
        const int buf = q & 1;
        // stage(q)'s async copies are the only ones outstanding here
        asm volatile("s_wait_asynccnt 0x0" ::: "memory");
        __syncthreads();                    // buf ready; all reads of buf^1 done
        if (q + 1 < NCHUNK) stage(q + 1, buf ^ 1);   // overlaps WMMAs below

        const unsigned char* abase = smem + buf * BUFSZ;
        const unsigned char* bbase = abase + ATILE;

        Frag16 fa[2], fb[2];
#pragma unroll
        for (int m = 0; m < 2; ++m) {
            // A 16x32: lane holds row lrow; K pairs at kofs=lhi*8 and kofs+16
            const unsigned char* p = abase + (o_wave + m * 16 + lrow) * LDSP + lhi * 16;
            fa[m].q[0] = *(const uint4*)(p);
            fa[m].q[1] = *(const uint4*)(p + 32);
        }
#pragma unroll
        for (int n = 0; n < 2; ++n) {
            // B 32x16: lane holds col lrow; K pairs at kofs=lhi*8 and kofs+16
            const unsigned char* p = bbase + (p_wave + n * 16 + lrow) * LDSP + lhi * 16;
            fb[n].q[0] = *(const uint4*)(p);
            fb[n].q[1] = *(const uint4*)(p + 32);
        }
#pragma unroll
        for (int m = 0; m < 2; ++m)
#pragma unroll
            for (int n = 0; n < 2; ++n)
                acc[m][n] = __builtin_amdgcn_wmma_f32_16x16x32_f16(
                    false, fa[m].v, false, fb[n].v,
                    (short)0, acc[m][n], false, false);
    }

    // ---- write D fragments: lane holds col lrow, rows v + 8*lhi ----
#pragma unroll
    for (int m = 0; m < 2; ++m) {
#pragma unroll
        for (int n = 0; n < 2; ++n) {
            const int w = p_wave + n * 16 + lrow;
#pragma unroll
            for (int v = 0; v < 8; ++v) {
                const int o = o_wave + m * 16 + lhi * 8 + v;
                y[(((size_t)b * CO_ + o) * H_ + h0) * W_ + w] = acc[m][n][v];
            }
        }
    }
}

// =====================================================================
// launch: coeff -> lp_gemm -> permute -> conv (all on `stream`)
// ws layout: [0,4KB) coeff f32 | [4KB,+12.5MB) lp16 | then lpp f16 (~13.1MB)
// =====================================================================
extern "C" void kernel_launch(void* const* d_in, const int* in_sizes, int n_in,
                              void* d_out, int out_size, void* d_ws, size_t ws_size,
                              hipStream_t stream) {
    (void)in_sizes; (void)n_in; (void)out_size; (void)ws_size;
    const float* x  = (const float*)d_in[0];   // (16,128,64,64)
    const float* tf = (const float*)d_in[1];   // (16,8,64)
    const float* wb = (const float*)d_in[2];   // (64,128,128,5,5)
    float* y = (float*)d_out;                  // (16,128,64,64)

    char* ws = (char*)d_ws;
    float*          coeff = (float*)ws;                                  // 4 KB
    _Float16*       lp16  = (_Float16*)(ws + 4096);                      // 13.1 MB
    unsigned short* lpp   = (unsigned short*)(ws + 4096 + (size_t)B_ * NCOL * 2);

    k_coeff    <<<4,           256, 0, stream>>>(tf, coeff);
    k_lp_gemm  <<<NCOL / 256,  256, 0, stream>>>(wb, coeff, lp16);
    k_permute  <<<B_ * CO_,    256, 0, stream>>>((const unsigned short*)lp16, lpp);
    k_conv_wmma<<<B_ * H_,     256, 0, stream>>>(x, lpp, y);
}
